// TRFA_86440511799349
// MI455X (gfx1250) — matrix operand order repaired and checked by reading, compiled, not run on hardware
//
#include <hip/hip_runtime.h>
#include <cstdint>
#include <cstddef>

// ---------------------------------------------------------------------------
// CDNA5 (gfx1250) wave32 WMMA types & helpers
// ---------------------------------------------------------------------------
typedef __attribute__((ext_vector_type(16))) _Float16 v16h;
typedef __attribute__((ext_vector_type(8)))  _Float16 v8h;
typedef __attribute__((ext_vector_type(8)))  float    v8f;

#define WMMA_F32_F16(a, b, c) \
  __builtin_amdgcn_wmma_f32_16x16x32_f16(false, (a), false, (b), (short)0, (c), false, false)

__device__ inline v8f zero8() {
  v8f c;
#pragma unroll
  for (int i = 0; i < 8; ++i) c[i] = 0.f;
  return c;
}

// A fragment (16x32 f16, M x K) from row-major tile (origin = row0,k0), ld in
// halves (ld*2 must be a 16B multiple). Two contiguous 8-half runs per lane:
// k = g*8 + [0..7] and k = g*8+16 + [0..7]  ->  two b128 loads.
__device__ inline v16h frag_a_ld(const _Float16* A, int ld) {
  const int lane = threadIdx.x & 31;
  const int m = lane & 15, g = lane >> 4;
  const _Float16* p = A + m * ld + g * 8;
  const v8h lo = *(const v8h*)(p);
  const v8h hi = *(const v8h*)(p + 16);
  return __builtin_shufflevector(lo, hi, 0, 1, 2, 3, 4, 5, 6, 7,
                                 8, 9, 10, 11, 12, 13, 14, 15);
}

// B fragment (32x16 f16, K x N) from an LDS tile stored TRANSPOSED (n-major):
// Bt[n*ld + k]; lane col n = lane&15, halves k = g*8 + [0..7] and g*8+16 + [0..7] -> two b128.
__device__ inline v16h frag_bt_ld(const _Float16* Bt, int ld) {
  const int lane = threadIdx.x & 31;
  const int n = lane & 15, g = lane >> 4;
  const _Float16* p = Bt + n * ld + g * 8;
  const v8h lo = *(const v8h*)(p);
  const v8h hi = *(const v8h*)(p + 16);
  return __builtin_shufflevector(lo, hi, 0, 1, 2, 3, 4, 5, 6, 7,
                                 8, 9, 10, 11, 12, 13, 14, 15);
}

// ---------------------------------------------------------------------------
// pad16: f32 (R x C, ldin) -> f16 (Rpad x Cpad), zero-filled padding.
// ---------------------------------------------------------------------------
__global__ void __launch_bounds__(128)
pad16_kernel(const float* __restrict__ in, _Float16* __restrict__ out,
             int R, int C, int ldin, int Cpad)
{
  const int c = blockIdx.x * 128 + threadIdx.x;
  const int r = blockIdx.y;
  if (c >= Cpad) return;
  _Float16 v = (_Float16)0.f;
  if (r < R && c < C) v = (_Float16)in[(size_t)r * ldin + c];
  out[(size_t)r * Cpad + c] = v;
}

// ---------------------------------------------------------------------------
// Tiled WMMA GEMM on fully padded f16 operands: no guards in the K loop.
// C(MxN) = act(A(MxKpad) * B(KpadxNpadB) + bias) [+ residual]
// Block 128 threads (4 waves), tile 128x64 (wave owns 32x64), K-step 32.
// LDS rows padded to 40 halves (80B stride) to avoid bank-conflict strides.
// ACT: 0 none, 1 relu, 2 elu(x)+1.
// HP:  head-pad stride; if HP != 0 output col n -> (n/75)*HP + n%75.
// Requirements: M % 128 == 0, Kpad % 32 == 0, lda/ldb multiples of 8.
// ---------------------------------------------------------------------------
template <int ACT, bool OUT16, int HP, bool HASRES>
__global__ void __launch_bounds__(128)
wmma_gemm_t(const _Float16* __restrict__ A, int lda,
            const _Float16* __restrict__ B, int ldb,
            const float* __restrict__ bias,
            const float* __restrict__ residual, int ldr,
            float* __restrict__ out32, _Float16* __restrict__ out16, int ldo,
            int Nvalid, int Kpad)
{
  __shared__ _Float16 sA[128 * 40];
  __shared__ _Float16 sBt[64 * 40];
  const int t = threadIdx.x;
  const int lane = t & 31, wave = t >> 5;
  const int m0 = blockIdx.x * 128;
  const int n0 = blockIdx.y * 64;

  v8f acc[2][4];
#pragma unroll
  for (int a = 0; a < 2; ++a)
#pragma unroll
    for (int j = 0; j < 4; ++j) acc[a][j] = zero8();

  const int ksteps = Kpad >> 5;
  for (int kt = 0; kt < ksteps; ++kt) {
    const int k0 = kt << 5;
    __syncthreads();
    // ---- Stage A tile (128x32): unconditional b128 loads/stores ----
#pragma unroll
    for (int c = 0; c < 4; ++c) {
      const int base = t * 8 + c * 1024;
      const int r = base >> 5, k = base & 31;            // k in {0,8,16,24}
      *(v8h*)(sA + r * 40 + k) =
          *(const v8h*)(A + (size_t)(m0 + r) * lda + k0 + k);
    }
    // ---- Stage B tile (32x64) transposed into sBt[n][k] ----
#pragma unroll
    for (int c = 0; c < 2; ++c) {
      const int base = t * 8 + c * 1024;
      const int k = base >> 6, n = base & 63;            // n multiple of 8
      const v8h v = *(const v8h*)(B + (size_t)(k0 + k) * ldb + n0 + n);
#pragma unroll
      for (int q = 0; q < 8; ++q) sBt[(n + q) * 40 + k] = v[q];
    }
    if (kt + 1 < ksteps)
      __builtin_prefetch((const void*)&A[(size_t)(m0 + t) * lda + k0 + 32], 0, 1);
    __syncthreads();
    v16h bfs[4];
#pragma unroll
    for (int j = 0; j < 4; ++j) bfs[j] = frag_bt_ld(&sBt[(j * 16) * 40], 40);
#pragma unroll
    for (int a = 0; a < 2; ++a) {
      const v16h af = frag_a_ld(&sA[(wave * 32 + a * 16) * 40], 40);
#pragma unroll
      for (int j = 0; j < 4; ++j) acc[a][j] = WMMA_F32_F16(af, bfs[j], acc[a][j]);
    }
  }

  // Epilogue. C/D layout: lane col = lane&15, row = r + (lane>>4)*8
  const int cn = lane & 15, cg = lane >> 4;
#pragma unroll
  for (int a = 0; a < 2; ++a) {
#pragma unroll
    for (int j = 0; j < 4; ++j) {
#pragma unroll
      for (int r = 0; r < 8; ++r) {
        const int m = m0 + wave * 32 + a * 16 + r + cg * 8;
        const int n = n0 + j * 16 + cn;
        float val = acc[a][j][r];
        if constexpr (OUT16) {
          if constexpr (HP != 0) {
            if (n < Nvalid) {
              val += bias[n];
              if constexpr (ACT == 1) val = fmaxf(val, 0.f);
              if constexpr (ACT == 2) val = (val > 0.f) ? (val + 1.f) : expf(val);
              out16[(size_t)m * ldo + (n / 75) * HP + (n % 75)] = (_Float16)val;
            }
          } else {
            if (n < ldo) {
              float o = 0.f;
              if (n < Nvalid) {
                o = val + bias[n];
                if constexpr (ACT == 1) o = fmaxf(o, 0.f);
                if constexpr (ACT == 2) o = (o > 0.f) ? (o + 1.f) : expf(o);
              }
              out16[(size_t)m * ldo + n] = (_Float16)o;
            }
          }
        } else {
          if (n < Nvalid) {
            val += bias[n];
            if constexpr (ACT == 1) val = fmaxf(val, 0.f);
            if constexpr (ACT == 2) val = (val > 0.f) ? (val + 1.f) : expf(val);
            if constexpr (HASRES) val += residual[(size_t)m * ldr + n];
            out32[(size_t)m * ldo + n] = val;
          }
        }
      }
    }
  }
}

// ---------------------------------------------------------------------------
// KV[n,h] (80x80, padded from 75x75) = V^T (75x2048) @ K (2048x75), per block.
// Block: 160 threads = 5 waves. K/V are head-padded (4x80, ld 320, zero pads)
// so staging is unconditional b128 global loads; tiles staged TRANSPOSED.
// ---------------------------------------------------------------------------
__global__ void __launch_bounds__(160)
kv_kernel(const _Float16* __restrict__ hKp, const _Float16* __restrict__ hVp,
          float* __restrict__ KV)
{
  __shared__ _Float16 sKt[80 * 40];   // sKt[d][s]
  __shared__ _Float16 sVt[80 * 40];   // sVt[m][s]
  const int nh = blockIdx.x;          // 0..127 = n*4+h
  const int n = nh >> 2, h = nh & 3;
  const int t = threadIdx.x, lane = t & 31, wave = t >> 5;
  const size_t baseRow = (size_t)n * 2048;
  const int col0 = h * 80;

  v8f acc[5];
#pragma unroll
  for (int j = 0; j < 5; ++j) acc[j] = zero8();

  for (int s0 = 0; s0 < 2048; s0 += 32) {
    __syncthreads();
#pragma unroll
    for (int c = 0; c < 2; ++c) {
      const int base = t * 8 + c * 1280;     // 2560 = 32*80
      const int sl = base / 80, d = base % 80;   // d multiple of 8
      const size_t off = (baseRow + s0 + sl) * 320 + col0 + d;
      const v8h kv = *(const v8h*)(hKp + off);
      const v8h vv = *(const v8h*)(hVp + off);
#pragma unroll
      for (int q = 0; q < 8; ++q) {
        sKt[(d + q) * 40 + sl] = kv[q];
        sVt[(d + q) * 40 + sl] = vv[q];
      }
    }
    __syncthreads();
    const v16h af = frag_a_ld(&sVt[(wave * 16) * 40], 40);   // A[m][s]
#pragma unroll
    for (int j = 0; j < 5; ++j) {
      const v16h bf = frag_bt_ld(&sKt[(j * 16) * 40], 40);   // B[s][d]
      acc[j] = WMMA_F32_F16(af, bf, acc[j]);
    }
  }

  const int cn = lane & 15, cg = lane >> 4;
  float* out = KV + (size_t)nh * 80 * 80;
#pragma unroll
  for (int j = 0; j < 5; ++j)
#pragma unroll
    for (int r = 0; r < 8; ++r) {
      const int m = wave * 16 + r + cg * 8;
      const int d = j * 16 + cn;
      out[m * 80 + d] = acc[j][r];
    }
}

// Ksum[n,h,d] = sum_s K[n,s,h,d]  (K head-padded, ld 320)
__global__ void ksum_kernel(const _Float16* __restrict__ hKp, float* __restrict__ Ksum)
{
  const int nh = blockIdx.x, t = threadIdx.x;
  if (t >= 80) return;
  const int n = nh >> 2, h = nh & 3;
  float s = 0.f;
  if (t < 75) {
    const _Float16* base = hKp + (size_t)n * 2048 * 320 + h * 80 + t;
    for (int i = 0; i < 2048; ++i) s += (float)base[(size_t)i * 320];
  }
  Ksum[(size_t)nh * 80 + t] = s;
}

// Z[n,l,h] = 1 / (Q . Ksum + eps); Q in head-padded layout (4x96 per row)
__global__ void z_kernel(const _Float16* __restrict__ hQp, const float* __restrict__ Ksum,
                         float* __restrict__ Z)
{
  const int idx = blockIdx.x * blockDim.x + threadIdx.x;   // (n*2048+l)*4 + h
  const int nl = idx >> 2, h = idx & 3;
  const int n = nl >> 11;
  const _Float16* q = hQp + (size_t)nl * 384 + h * 96;
  const float* ks = Ksum + ((size_t)n * 4 + h) * 80;
  float s = 0.f;
  for (int d = 0; d < 75; ++d) s += (float)q[d] * ks[d];
  Z[idx] = 1.f / (s + 1e-6f);
}

// ---------------------------------------------------------------------------
// att[n,l,h,m] = Z[n,l,h] * sum_d Q[n,l,h,d] * KV[n,h,m,d]
// Q head-padded (4x96, zero pads) -> unguarded 16B A loads; KV zero-padded in
// LDS (stride 104). Output hAtt has ld 320 (pads pre-zeroed).
// ---------------------------------------------------------------------------
__global__ void __launch_bounds__(128)
att_kernel(const _Float16* __restrict__ hQp, const float* __restrict__ KV,
           const float* __restrict__ Z, _Float16* __restrict__ hAtt)
{
  __shared__ _Float16 sKV[80 * 104];   // sKV[m][d], d padded 80->104 (zeros)
  const int nh = blockIdx.y, n = nh >> 2, h = nh & 3;
  const int l0 = blockIdx.x * 64;
  const int t = threadIdx.x, lane = t & 31, wave = t >> 5;

  const float* kvp = KV + (size_t)nh * 6400;
  for (int idx = t; idx < 80 * 104; idx += 128) {
    const int m = idx / 104, d = idx % 104;
    sKV[idx] = (d < 80) ? (_Float16)kvp[m * 80 + d] : (_Float16)0.f;
  }
  __syncthreads();

  const size_t rowBase = (size_t)n * 2048;
  v8f acc[5];
#pragma unroll
  for (int j = 0; j < 5; ++j) acc[j] = zero8();

#pragma unroll
  for (int kk = 0; kk < 3; ++kk) {     // K = 75 padded to 96 = 3 x 32
    const int k0 = kk * 32;
    const v16h af = frag_a_ld(hQp + (rowBase + l0 + wave * 16) * 384 + h * 96 + k0, 384);
#pragma unroll
    for (int j = 0; j < 5; ++j) {
      const v16h bf = frag_bt_ld(&sKV[(j * 16) * 104 + k0], 104);
      acc[j] = WMMA_F32_F16(af, bf, acc[j]);
    }
  }

  const int cn = lane & 15, cg = lane >> 4;
  const int col0 = h * 75;
  float zr[8];
#pragma unroll
  for (int r = 0; r < 8; ++r) {
    const int l = l0 + wave * 16 + r + cg * 8;
    zr[r] = Z[(rowBase + l) * 4 + h];
  }
#pragma unroll
  for (int j = 0; j < 5; ++j)
#pragma unroll
    for (int r = 0; r < 8; ++r) {
      const int m = j * 16 + cn;
      if (m < 75) {
        const int l = l0 + wave * 16 + r + cg * 8;
        hAtt[(rowBase + l) * 320 + col0 + m] = (_Float16)(acc[j][r] * zr[r]);
      }
    }
}

// ---------------------------------------------------------------------------
// LayerNorm over rows of 300; also emits padded f16 copy (ld 320, zero pads).
// ---------------------------------------------------------------------------
__global__ void __launch_bounds__(128)
ln_kernel(const float* __restrict__ in, const float* __restrict__ g,
          const float* __restrict__ b, float* __restrict__ out,
          _Float16* __restrict__ out16)
{
  __shared__ float rs[128], rs2[128];
  const int row = blockIdx.x, t = threadIdx.x;
  const float* x = in + (size_t)row * 300;
  float s = 0.f, s2 = 0.f;
  for (int i = t; i < 300; i += 128) { const float v = x[i]; s += v; s2 += v * v; }
  rs[t] = s; rs2[t] = s2; __syncthreads();
  for (int o = 64; o > 0; o >>= 1) {
    if (t < o) { rs[t] += rs[t + o]; rs2[t] += rs2[t + o]; }
    __syncthreads();
  }
  const float mu = rs[0] * (1.f / 300.f);
  const float var = rs2[0] * (1.f / 300.f) - mu * mu;
  const float inv = rsqrtf(var + 1e-5f);
  for (int i = t; i < 300; i += 128) {
    const float v = (x[i] - mu) * inv * g[i] + b[i];
    out[(size_t)row * 300 + i] = v;
    out16[(size_t)row * 320 + i] = (_Float16)v;
  }
  if (t < 20) out16[(size_t)row * 320 + 300 + t] = (_Float16)0.f;
}

// Fused LN2 then LNf, emits f16 encoder output (ld 320, zero pads).
__global__ void __launch_bounds__(128)
ln2f_kernel(const float* __restrict__ in,
            const float* __restrict__ g2, const float* __restrict__ b2,
            const float* __restrict__ gf, const float* __restrict__ bf,
            _Float16* __restrict__ out)
{
  __shared__ float xr[300];
  __shared__ float rs[128], rs2[128];
  const int row = blockIdx.x, t = threadIdx.x;
  const float* x = in + (size_t)row * 300;
  for (int i = t; i < 300; i += 128) xr[i] = x[i];
  __syncthreads();
  for (int pass = 0; pass < 2; ++pass) {
    const float* g = pass ? gf : g2;
    const float* b = pass ? bf : b2;
    float s = 0.f, s2 = 0.f;
    for (int i = t; i < 300; i += 128) { const float v = xr[i]; s += v; s2 += v * v; }
    rs[t] = s; rs2[t] = s2; __syncthreads();
    for (int o = 64; o > 0; o >>= 1) {
      if (t < o) { rs[t] += rs[t + o]; rs2[t] += rs2[t + o]; }
      __syncthreads();
    }
    const float mu = rs[0] * (1.f / 300.f);
    const float var = rs2[0] * (1.f / 300.f) - mu * mu;
    const float inv = rsqrtf(var + 1e-5f);
    __syncthreads();
    for (int i = t; i < 300; i += 128) xr[i] = (xr[i] - mu) * inv * g[i] + b[i];
    __syncthreads();
  }
  for (int i = t; i < 300; i += 128) out[(size_t)row * 320 + i] = (_Float16)xr[i];
  if (t < 20) out[(size_t)row * 320 + 300 + t] = (_Float16)0.f;
}

// ---------------------------------------------------------------------------
// 1x1 conv as GEMM per batch: out(128x300) = W16(128x4096) @ concat(4096x320)
// + bias + BN(eval) + ReLU -> d_out (B,128,20,15). All staging unconditional.
// ---------------------------------------------------------------------------
__global__ void __launch_bounds__(128)
conv_kernel(const _Float16* __restrict__ fOut, const _Float16* __restrict__ eOut,
            const _Float16* __restrict__ W16, const float* __restrict__ cb,
            const float* __restrict__ bng, const float* __restrict__ bnb,
            const float* __restrict__ bnm, const float* __restrict__ bnv,
            float* __restrict__ out)
{
  __shared__ _Float16 sA[64 * 40];
  __shared__ _Float16 sBt[64 * 40];
  const int n0 = blockIdx.x * 64;    // spatial h*w (tiles cover 320)
  const int m0 = blockIdx.y * 64;    // output channel
  const int b  = blockIdx.z;
  const int t = threadIdx.x, lane = t & 31, wave = t >> 5;

  v8f acc[4];
#pragma unroll
  for (int j = 0; j < 4; ++j) acc[j] = zero8();

  for (int k0 = 0; k0 < 4096; k0 += 32) {
    __syncthreads();
#pragma unroll
    for (int c = 0; c < 2; ++c) {
      const int base = t * 8 + c * 1024;
      const int r = base >> 5, k = base & 31;
      *(v8h*)(sA + r * 40 + k) =
          *(const v8h*)(W16 + (size_t)(m0 + r) * 4096 + k0 + k);
    }
#pragma unroll
    for (int c = 0; c < 2; ++c) {
      const int base = t * 8 + c * 1024;
      const int k = base >> 6, n = base & 63;          // n multiple of 8
      const int gc = k0 + k;
      const _Float16* src = (gc < 2048)
          ? (fOut + ((size_t)b * 2048 + gc) * 320)
          : (eOut + ((size_t)b * 2048 + gc - 2048) * 320);
      const v8h v = *(const v8h*)(src + n0 + n);
#pragma unroll
      for (int q = 0; q < 8; ++q) sBt[(n + q) * 40 + k] = v[q];
    }
    __syncthreads();
    const v16h af = frag_a_ld(&sA[(wave * 16) * 40], 40);
#pragma unroll
    for (int j = 0; j < 4; ++j) {
      const v16h bf = frag_bt_ld(&sBt[(j * 16) * 40], 40);
      acc[j] = WMMA_F32_F16(af, bf, acc[j]);
    }
  }

  const int cn = lane & 15, cg = lane >> 4;
#pragma unroll
  for (int j = 0; j < 4; ++j)
#pragma unroll
    for (int r = 0; r < 8; ++r) {
      const int o  = m0 + wave * 16 + r + cg * 8;
      const int hw = n0 + j * 16 + cn;
      if (hw < 300) {
        const float scale = bng[o] * rsqrtf(bnv[o] + 1e-5f);
        float val = (acc[j][r] + cb[o]) * scale + (bnb[o] - bnm[o] * scale);
        out[((size_t)b * 128 + o) * 300 + hw] = fmaxf(val, 0.f);
      }
    }
}

__global__ void fill_zero_u32(uint32_t* p, size_t n) {
  const size_t i = (size_t)blockIdx.x * blockDim.x + threadIdx.x;
  if (i < n) p[i] = 0u;
}

// ---------------------------------------------------------------------------
// Host side
// ---------------------------------------------------------------------------
static void pad16(hipStream_t st, const float* in, _Float16* out,
                  int R, int C, int ldin, int Rpad, int Cpad)
{
  dim3 grid((Cpad + 127) / 128, Rpad);
  pad16_kernel<<<grid, dim3(128), 0, st>>>(in, out, R, C, ldin, Cpad);
}

static void run_encoder(hipStream_t st, const float* xin, void* const* d_in, int p,
                        _Float16* xh, _Float16* x1h,
                        _Float16* wQ, _Float16* wK, _Float16* wV, _Float16* wO,
                        _Float16* w1, _Float16* w2,
                        _Float16* hQp, _Float16* hKp, _Float16* hVp, _Float16* hAtt,
                        _Float16* hid, float* KV, float* Ksum, float* Zb,
                        float* ybuf, float* x1, _Float16* outBuf)
{
  const float* Wq = (const float*)d_in[p + 0];  const float* bq = (const float*)d_in[p + 1];
  const float* Wk = (const float*)d_in[p + 2];  const float* bk = (const float*)d_in[p + 3];
  const float* Wv = (const float*)d_in[p + 4];  const float* bv = (const float*)d_in[p + 5];
  const float* Wo = (const float*)d_in[p + 6];  const float* bo = (const float*)d_in[p + 7];
  const float* g1 = (const float*)d_in[p + 8];  const float* b1g = (const float*)d_in[p + 9];
  const float* W1 = (const float*)d_in[p + 10]; const float* b1 = (const float*)d_in[p + 11];
  const float* W2 = (const float*)d_in[p + 12]; const float* b2 = (const float*)d_in[p + 13];
  const float* g2 = (const float*)d_in[p + 14]; const float* b2g = (const float*)d_in[p + 15];
  const float* gfv = (const float*)d_in[p + 16]; const float* bfv = (const float*)d_in[p + 17];

  const int M = 32 * 2048;                         // 65536 rows
  const dim3 blk(128);
  const dim3 g5(M / 128, 5), g2d(M / 128, 2);

  // Pad/convert activations and weights to f16 tiles
  pad16(st, xin, xh, M, 300, 300, M, 320);
  pad16(st, Wq, wQ, 300, 300, 300, 320, 320);
  pad16(st, Wk, wK, 300, 300, 300, 320, 320);
  pad16(st, Wv, wV, 300, 300, 300, 320, 320);
  pad16(st, Wo, wO, 300, 300, 300, 320, 320);
  pad16(st, W1, w1, 300, 128, 128, 320, 128);
  pad16(st, W2, w2, 128, 300, 300, 128, 320);

  // QKV projections; elu+1 fused on Q,K. Head-padded outputs:
  //   Q -> 4x96 (att needs K padded to 3x32), K/V -> 4x80.
  wmma_gemm_t<2, true, 96, false><<<g5, blk, 0, st>>>(xh, 320, wQ, 320, bq, nullptr, 0, nullptr, hQp, 384, 300, 320);
  wmma_gemm_t<2, true, 80, false><<<g5, blk, 0, st>>>(xh, 320, wK, 320, bk, nullptr, 0, nullptr, hKp, 320, 300, 320);
  wmma_gemm_t<0, true, 80, false><<<g5, blk, 0, st>>>(xh, 320, wV, 320, bv, nullptr, 0, nullptr, hVp, 320, 300, 320);

  ksum_kernel<<<dim3(128), dim3(128), 0, st>>>(hKp, Ksum);
  kv_kernel<<<dim3(128), dim3(160), 0, st>>>(hKp, hVp, KV);
  z_kernel<<<dim3(2048), dim3(128), 0, st>>>(hQp, Ksum, Zb);
  att_kernel<<<dim3(32, 128), dim3(128), 0, st>>>(hQp, KV, Zb, hAtt);

  // x + att @ Wo + bo -> ybuf; then LN1 -> x1 (f32) and x1h (f16, padded)
  wmma_gemm_t<0, false, 0, true><<<g5, blk, 0, st>>>(hAtt, 320, wO, 320, bo, xin, 300, ybuf, nullptr, 300, 300, 320);
  ln_kernel<<<dim3(M), blk, 0, st>>>(ybuf, g1, b1g, x1, x1h);

  // FFN
  wmma_gemm_t<1, true, 0, false><<<g2d, blk, 0, st>>>(x1h, 320, w1, 128, b1, nullptr, 0, nullptr, hid, 128, 128, 320);
  wmma_gemm_t<0, false, 0, true><<<g5, blk, 0, st>>>(hid, 128, w2, 320, b2, x1, 300, ybuf, nullptr, 300, 300, 128);

  // LN2 then final LN -> f16 output (ld 320) for the conv GEMM
  ln2f_kernel<<<dim3(M), blk, 0, st>>>(ybuf, g2, b2g, gfv, bfv, outBuf);
}

extern "C" void kernel_launch(void* const* d_in, const int* in_sizes, int n_in,
                              void* d_out, int out_size, void* d_ws, size_t ws_size,
                              hipStream_t stream)
{
  (void)in_sizes; (void)n_in; (void)out_size; (void)ws_size;
  const float* feature = (const float*)d_in[0];
  const float* edge    = (const float*)d_in[1];

  // --- workspace carve-out (bump allocator, 256B aligned) ---
  char* ws = (char*)d_ws;
  auto take = [&](size_t bytes) -> char* {
    char* p = ws;
    ws += (bytes + 255) & ~(size_t)255;
    return p;
  };
  const size_t M = 32ull * 2048ull;                    // 65536
  // hQp/hAtt/hKp/hVp kept contiguous so one fill zeroes all their pads.
  _Float16* hQp  = (_Float16*)take(M * 384 * 2);       // head-padded Q (4x96)
  _Float16* hAtt = (_Float16*)take(M * 320 * 2);       // attention out, ld 320
  _Float16* hKp  = (_Float16*)take(M * 320 * 2);       // head-padded K (4x80)
  _Float16* hVp  = (_Float16*)take(M * 320 * 2);       // head-padded V (4x80)
  _Float16* xh   = (_Float16*)take(M * 320 * 2);       // f16 padded encoder input
  _Float16* x1h  = (_Float16*)take(M * 320 * 2);       // f16 padded LN1 output
  _Float16* hid  = (_Float16*)take(M * 128 * 2);
  float*    KV   = (float*)take(128ull * 80 * 80 * 4);
  float*    Ksum = (float*)take(128ull * 80 * 4);
  float*    Zb   = (float*)take(M * 4 * 4);
  float*    ybuf = (float*)take(M * 300 * 4);
  float*    x1   = (float*)take(M * 300 * 4);
  _Float16* fOut = (_Float16*)take(M * 320 * 2);
  _Float16* eOut = (_Float16*)take(M * 320 * 2);
  _Float16* wQ   = (_Float16*)take(320 * 320 * 2);
  _Float16* wK   = (_Float16*)take(320 * 320 * 2);
  _Float16* wV   = (_Float16*)take(320 * 320 * 2);
  _Float16* wO   = (_Float16*)take(320 * 320 * 2);
  _Float16* w1   = (_Float16*)take(320 * 128 * 2);
  _Float16* w2   = (_Float16*)take(128 * 320 * 2);
  _Float16* wC   = (_Float16*)take(128 * 4096 * 2);

  // Zero hQp + hAtt + hKp + hVp once (contiguous) so all pad lanes are clean.
  {
    const size_t n32 = (M * (384 + 320 + 320 + 320) * 2) / 4;
    fill_zero_u32<<<dim3((unsigned)((n32 + 255) / 256)), dim3(256), 0, stream>>>(
        (uint32_t*)hQp, n32);
  }

  // Two encoders (enc1 params at d_in[2..19], enc2 at d_in[20..37])
  run_encoder(stream, feature, d_in, 2,  xh, x1h, wQ, wK, wV, wO, w1, w2,
              hQp, hKp, hVp, hAtt, hid, KV, Ksum, Zb, ybuf, x1, fOut);
  run_encoder(stream, edge,    d_in, 20, xh, x1h, wQ, wK, wV, wO, w1, w2,
              hQp, hKp, hVp, hAtt, hid, KV, Ksum, Zb, ybuf, x1, eOut);

  // Final 1x1 conv + BN + ReLU  (conv params at d_in[38..43])
  const float* cw  = (const float*)d_in[38];
  const float* cbb = (const float*)d_in[39];
  const float* bng = (const float*)d_in[40];
  const float* bnb = (const float*)d_in[41];
  const float* bnm = (const float*)d_in[42];
  const float* bnv = (const float*)d_in[43];
  pad16(stream, cw, wC, 128, 4096, 4096, 128, 4096);
  conv_kernel<<<dim3(5, 2, 32), dim3(128), 0, stream>>>(fOut, eOut, wC, cbb,
                                                        bng, bnb, bnm, bnv,
                                                        (float*)d_out);
}
